// PretrainModel_62672162783931
// MI455X (gfx1250) — matrix operand order repaired and checked by reading, not compiled, not run on hardware
//
#include <hip/hip_runtime.h>
#include <hip/hip_bf16.h>
#include <math.h>

// ---------------- problem constants (from reference) ----------------
#define B_     64
#define N_TOK  64
#define M_TOK  256
#define C_     256
#define H_     8
#define HD     32                  // C_/H_
#define TH_    50
#define TF_    60
#define MODES_ 6
#define T_     (2*N_TOK + M_TOK)   // 384
#define C3     (3*C_)              // 768
#define CM     (4*C_)              // 1024 (MLP hidden)
#define OUT_PER_B (M_TOK*40 + N_TOK*MODES_*TH_*2 + N_TOK*MODES_*TF_*2) // 94720
#define AW     8                   // waves per attention workgroup

typedef __attribute__((ext_vector_type(16))) _Float16 v16h;
typedef __attribute__((ext_vector_type(8)))  _Float16 v8h;
typedef __attribute__((ext_vector_type(8)))  float    v8f;
typedef unsigned int u32x4 __attribute__((ext_vector_type(4)));
typedef int          i32x8 __attribute__((ext_vector_type(8)));
typedef int          i32x4 __attribute__((ext_vector_type(4)));

#if __has_builtin(__builtin_amdgcn_tensor_load_to_lds)
#define HAVE_TDM 1
#else
#define HAVE_TDM 0
#endif

// ---------------- WMMA helper (CDNA5 16x16x32 f16 -> f32 acc) ----------------
__device__ __forceinline__ v8f wmma16(v16h a, v16h b, v8f c) {
  return __builtin_amdgcn_wmma_f32_16x16x32_f16(false, a, false, b, (short)0, c,
                                                false, false);
}

// A fragment 16x32 (MxK) from row-major f16, leading dim lda (halves).
// lane<16 -> M=lane, elements 0..7 = K koff..+7, 8..15 = K koff+16..+23; koff=8*(lane>=16)
__device__ __forceinline__ v16h load_a16(const _Float16* __restrict__ A, int lda, int lane) {
  const _Float16* row = A + (long)(lane & 15) * lda + ((lane >> 4) << 3);
  v8h lo = *(const v8h*)(row);
  v8h hi = *(const v8h*)(row + 16);
  return __builtin_shufflevector(lo, hi, 0,1,2,3,4,5,6,7,8,9,10,11,12,13,14,15);
}

// B fragment 32x16 (KxN) from a K-contiguous, N-strided panel P[n][k] (ld = ldw):
// used for weights W[n][k], K tiles K[key][d], and the transposed V panel V^T[d][key].
// lane<16 -> N=lane, elements 0..7 = K koff..+7, 8..15 = K koff+16..+23; koff=8*(lane>=16)
__device__ __forceinline__ v16h load_b16_wt(const _Float16* __restrict__ W, int ldw,
                                            int nvalid, int lane) {
  int n = lane & 15;
  if (n >= nvalid) { v16h z = {}; return z; }
  const _Float16* row = W + (long)n * ldw + ((lane >> 4) << 3);
  v8h lo = *(const v8h*)(row);
  v8h hi = *(const v8h*)(row + 16);
  return __builtin_shufflevector(lo, hi, 0,1,2,3,4,5,6,7,8,9,10,11,12,13,14,15);
}

// ---------------- TDM: 2-D tile global -> LDS (h rows of w f16, row stride) --
// w,h <= 65535 (tile_dim fields are 16-bit); stride in f16 elements.
__device__ __forceinline__ void tdm_copy_2d(const void* gptr, unsigned lds_off,
                                            unsigned w, unsigned h, unsigned stride) {
#if HAVE_TDM
  unsigned long long ga = (unsigned long long)gptr;
  u32x4 g0;
  g0[0] = 1u;                                        // count=1 (valid), user mode
  g0[1] = lds_off;                                   // lds_addr (bytes)
  g0[2] = (unsigned)(ga & 0xffffffffu);              // global_addr[31:0]
  g0[3] = (unsigned)((ga >> 32) & 0x01ffffffu)       // global_addr[56:32]
        | 0x80000000u;                               // type = 2 ("image")
  i32x8 g1;
  g1[0] = 0x00010000;                                // data_size=1 (2B), mask=0
  g1[1] = (int)((w & 0xffffu) << 16);                // tensor_dim0[15:0]  @63:48
  g1[2] = (int)(((w >> 16) & 0xffffu)                // tensor_dim0[31:16] @79:64
        | ((h & 0xffffu) << 16));                    // tensor_dim1[15:0]  @95:80
  g1[3] = (int)(((h >> 16) & 0xffffu)                // tensor_dim1[31:16] @111:96
        | ((w & 0xffffu) << 16));                    // tile_dim0          @127:112
  g1[4] = (int)(h & 0xffffu);                        // tile_dim1          @143:128
  g1[5] = (int)stride;                               // tensor_dim0_stride[31:0]
  g1[6] = 0;
  g1[7] = 0;
  i32x4 z4 = {0, 0, 0, 0};
#if defined(__clang_major__) && (__clang_major__ >= 23)
  i32x8 z8 = {0, 0, 0, 0, 0, 0, 0, 0};
  __builtin_amdgcn_tensor_load_to_lds(g0, g1, z4, z4, z8, 0);
#else
  __builtin_amdgcn_tensor_load_to_lds(g0, g1, z4, z4, 0);
#endif
  __builtin_amdgcn_s_wait_tensorcnt((short)0);
#else
  (void)gptr; (void)lds_off; (void)w; (void)h; (void)stride;
#endif
}

// ---------------- WMMA GEMM with TDM-staged A panel -------------------------
// Block = 4 waves: one 32-row A panel staged once into LDS; 4 consecutive
// 32-col tiles (one per wave) -> 32x128 output per block.
template <typename OT>
__global__ void __launch_bounds__(128) gemm_wmma_kernel(
    const _Float16* __restrict__ A, const _Float16* __restrict__ W,
    const float* __restrict__ bias, const float* __restrict__ resid,
    OT* __restrict__ Out, int N, int K,
    int a_T, int a_tok0, int tokcount,
    long o_bstride, int o_off, int o_ldo, int act)
{
  extern __shared__ __align__(16) _Float16 apanel[];   // 32 * K halves
  const int lane = threadIdx.x & 31;
  const int wv   = threadIdx.x >> 5;
  const int ntn   = (N + 31) >> 5;
  const int nblkn = (ntn + 3) >> 2;
  const int mt = blockIdx.x / nblkn;
  const int ng = blockIdx.x % nblkn;
  const int nt = ng * 4 + wv;

  const int mr = mt << 5;
  const int b  = mr / tokcount;
  const int t0 = mr % tokcount;
  const _Float16* A0 = A + (long)(b * a_T + a_tok0 + t0) * K;

  const int halves = 32 * K;
#if HAVE_TDM
  if (wv == 0)  // TDM ignores EXEC; issue once from wave 0, wait TENSORcnt
    tdm_copy_2d(A0, (unsigned)(size_t)apanel, (unsigned)halves, 1, (unsigned)halves);
#else
  for (int i = threadIdx.x; i < (halves >> 3); i += 128)
    ((v8h*)apanel)[i] = ((const v8h*)A0)[i];
#endif
  __syncthreads();   // all 4 waves reach this (even edge waves with nt >= ntn)

  if (nt >= ntn) return;
  const _Float16* W0 = W + (long)(nt << 5) * K;
  const _Float16* W1 = W0 + (long)16 * K;
  const int nv0 = N - (nt << 5);
  const int nv1 = nv0 - 16;

  v8f a00 = {}, a01 = {}, a10 = {}, a11 = {};
  for (int kc = 0; kc < K; kc += 32) {
    __builtin_prefetch(W0 + (long)(lane & 15) * K + kc + 32, 0, 1);
    v16h fa0 = load_a16(apanel + kc, K, lane);            // from LDS
    v16h fa1 = load_a16(apanel + 16 * K + kc, K, lane);   // from LDS
    v16h fb0 = load_b16_wt(W0 + kc, K, nv0, lane);
    v16h fb1 = load_b16_wt(W1 + kc, K, nv1, lane);
    a00 = wmma16(fa0, fb0, a00);
    a01 = wmma16(fa0, fb1, a01);
    a10 = wmma16(fa1, fb0, a10);
    a11 = wmma16(fa1, fb1, a11);
  }

  const int col16  = lane & 15;
  const int rowoff = (lane >> 4) << 3;
#pragma unroll
  for (int j = 0; j < 2; ++j) {
    int n = (nt << 5) + j * 16 + col16;
    if (n >= N) continue;
    float bv = bias ? bias[n] : 0.0f;
#pragma unroll
    for (int i = 0; i < 2; ++i) {
      const v8f acc = (i == 0) ? (j == 0 ? a00 : a01) : (j == 0 ? a10 : a11);
#pragma unroll
      for (int r = 0; r < 8; ++r) {
        int t = t0 + i * 16 + r + rowoff;
        long off = (long)b * o_bstride + o_off + (long)t * o_ldo + n;
        float v = acc[r] + bv;
        if (act == 1) v = 0.5f * v * (1.0f + erff(v * 0.70710678118f));
        if (resid) v += resid[off];
        Out[off] = (OT)v;
      }
    }
  }
}

// ---------------- fused MHA: one 8-wave workgroup per (b, h) -----------------
// K panel (384x32, row-major) TDM-staged; V panel staged TRANSPOSED (32x384)
// so AV B-fragments are two 16-byte ds loads; 8 waves x 3 q-tiles each.
__global__ void __launch_bounds__(256) attn_kernel(
    const _Float16* __restrict__ qkv,  // (B,T,3C) f16: [q|k|v]
    const int*      __restrict__ pad,  // (B,T) 1 = masked
    _Float16* __restrict__ out)        // (B,T,C) f16
{
  __shared__ __align__(16) _Float16 kpan[T_ * HD];        // 24 KB (key, d)
  __shared__ __align__(16) _Float16 vpanT[HD * T_];       // 24 KB (d, key)
  __shared__ __align__(16) _Float16 probs[AW][16 * T_];   // 96 KB
  const int lane = threadIdx.x & 31;
  const int wv   = threadIdx.x >> 5;
  const int h = blockIdx.x % H_;
  const int b = blockIdx.x / H_;

  const _Float16* base = qkv + (long)b * T_ * C3;
  const _Float16* qb = base + h * HD;
  const _Float16* kb = base + C_ + h * HD;
  const _Float16* vb = base + 2 * C_ + h * HD;

#if HAVE_TDM
  if (wv == 0)   // K panel: row-major bulk tile via the Tensor Data Mover
    tdm_copy_2d(kb, (unsigned)(size_t)kpan, HD, T_, C3);
#else
  for (int t = threadIdx.x; t < T_; t += 256) {
    const v8h* sk = (const v8h*)(kb + (long)t * C3);
    v8h* dk = (v8h*)(kpan + t * HD);
    dk[0] = sk[0]; dk[1] = sk[1]; dk[2] = sk[2]; dk[3] = sk[3];
  }
#endif
  // V panel: cooperative transposed staging (one contiguous 64B row read per
  // thread, scattered into LDS columns); amortized over 24 q tiles.
  for (int t = threadIdx.x; t < T_; t += 256) {
    const v8h* sv = (const v8h*)(vb + (long)t * C3);
    v8h r0 = sv[0], r1 = sv[1], r2 = sv[2], r3 = sv[3];
#pragma unroll
    for (int d = 0; d < 8; ++d) {
      vpanT[(d)      * T_ + t] = r0[d];
      vpanT[(d + 8)  * T_ + t] = r1[d];
      vpanT[(d + 16) * T_ + t] = r2[d];
      vpanT[(d + 24) * T_ + t] = r3[d];
    }
  }
  __syncthreads();

  const int col16  = lane & 15;
  const int rowoff = (lane >> 4) << 3;
  const float scale = 0.17677669529663687f;        // 1/sqrt(32)
  _Float16* sch = probs[wv];

  for (int qi = 0; qi < T_ / 16 / AW; ++qi) {      // 3 q tiles per wave
    const int qt = wv * (T_ / 16 / AW) + qi;
    v16h qa = load_a16(qb + (long)(qt * 16) * C3, C3, lane);

    // scores: hd=32 == one WMMA K-chunk; K panel from LDS
    for (int kt = 0; kt < T_ / 16; ++kt) {
      v16h kB = load_b16_wt(kpan + kt * 16 * HD, HD, 16, lane);
      v8f s = {};
      s = wmma16(qa, kB, s);
      int key = kt * 16 + col16;
      bool msk = pad[b * T_ + key] != 0;
#pragma unroll
      for (int r = 0; r < 8; ++r)
        sch[(r + rowoff) * T_ + key] =
            msk ? (_Float16)(-60000.0f) : (_Float16)(s[r] * scale);
    }
    __syncthreads();   // uniform trip count across all 8 waves

    // row softmax in f32 (lanes 0..15 own one row each)
    if (lane < 16) {
      _Float16* row = &sch[lane * T_];
      float mx = -3.4e38f;
      for (int j = 0; j < T_; ++j) mx = fmaxf(mx, (float)row[j]);
      float sum = 0.0f;
      for (int j = 0; j < T_; ++j) sum += __expf((float)row[j] - mx);
      float inv = 1.0f / sum;
      for (int j = 0; j < T_; ++j)
        row[j] = (_Float16)(__expf((float)row[j] - mx) * inv);
    }
    __syncthreads();

    // o = probs @ V : 12 K-chunks x two N halves; V^T panel -> vector ds loads
    v8f o0 = {}, o1 = {};
    for (int kc = 0; kc < T_; kc += 32) {
      v16h p  = load_a16(&sch[kc], T_, lane);
      v16h v0 = load_b16_wt(vpanT + kc, T_, 16, lane);            // d 0..15
      v16h v1 = load_b16_wt(vpanT + 16 * T_ + kc, T_, 16, lane);  // d 16..31
      o0 = wmma16(p, v0, o0);
      o1 = wmma16(p, v1, o1);
    }
#pragma unroll
    for (int r = 0; r < 8; ++r) {
      int t = qt * 16 + r + rowoff;
      long off = ((long)b * T_ + t) * C_ + h * HD;
      out[off + col16]      = (_Float16)o0[r];
      out[off + 16 + col16] = (_Float16)o1[r];
    }
  }
}

// ---------------- LayerNorm: one wave32 per token, f32 in, f16 out ----------
__global__ void __launch_bounds__(256) layernorm_kernel(
    const float* __restrict__ X, const float* __restrict__ g,
    const float* __restrict__ be, _Float16* __restrict__ Out, int ntok)
{
  int wid  = (blockIdx.x * blockDim.x + threadIdx.x) >> 5;
  int lane = threadIdx.x & 31;
  if (wid >= ntok) return;
  const float* x = X + (long)wid * C_;
  float4 a = ((const float4*)x)[lane * 2];
  float4 c = ((const float4*)x)[lane * 2 + 1];
  float v[8] = {a.x, a.y, a.z, a.w, c.x, c.y, c.z, c.w};
  float s = 0.0f, q = 0.0f;
#pragma unroll
  for (int i = 0; i < 8; ++i) { s += v[i]; q += v[i] * v[i]; }
  for (int off = 16; off > 0; off >>= 1) {
    s += __shfl_xor(s, off);
    q += __shfl_xor(q, off);
  }
  float mu  = s * (1.0f / C_);
  float var = q * (1.0f / C_) - mu * mu;
  float inv = rsqrtf(var + 1e-5f);
  _Float16* o = Out + (long)wid * C_;
  int c0 = lane * 8;
#pragma unroll
  for (int i = 0; i < 8; ++i)
    o[c0 + i] = (_Float16)((v[i] - mu) * inv * g[c0 + i] + be[c0 + i]);
}

// ---------------- small elementwise kernels ----------------
__global__ void cvt_f16_kernel(const float* __restrict__ s, _Float16* __restrict__ d, long n) {
  long i = (long)blockIdx.x * 256 + threadIdx.x;
  if (i < n) d[i] = (_Float16)s[i];
}

__global__ void build_x_kernel(const float* __restrict__ hist, const float* __restrict__ fut,
                               const float* __restrict__ lane, float* __restrict__ x)
{
  long i = (long)blockIdx.x * blockDim.x + threadIdx.x;
  if (i >= (long)B_ * T_ * C_) return;
  int c = (int)(i % C_); long bt = i / C_;
  int t = (int)(bt % T_); int b = (int)(bt / T_);
  float v;
  if (t < N_TOK)          v = hist[((long)b * N_TOK + t) * C_ + c];
  else if (t < 2 * N_TOK) v = fut [((long)b * N_TOK + (t - N_TOK)) * C_ + c];
  else                    v = lane[((long)b * M_TOK + (t - 2 * N_TOK)) * C_ + c];
  x[i] = v;
}

__global__ void enc_pad_kernel(const int* __restrict__ hk, const int* __restrict__ fk,
                               const int* __restrict__ lk, int* __restrict__ pad)
{
  int i = blockIdx.x * blockDim.x + threadIdx.x;
  if (i >= B_ * T_) return;
  int t = i % T_, b = i / T_, keep;
  if (t < N_TOK)          keep = hk[b * N_TOK + t];
  else if (t < 2 * N_TOK) keep = fk[b * N_TOK + t - N_TOK];
  else                    keep = lk[b * M_TOK + t - 2 * N_TOK];
  pad[i] = keep ? 0 : 1;
}

__global__ void dec_pad_kernel(const int* __restrict__ hm, const int* __restrict__ fm,
                               const int* __restrict__ lp, int* __restrict__ pad)
{
  int i = blockIdx.x * blockDim.x + threadIdx.x;
  if (i >= B_ * T_) return;
  int t = i % T_, b = i / T_, p;
  if (t < N_TOK) {
    int any = 0;
    for (int k = 0; k < TH_; ++k) any |= hm[((long)b * N_TOK + t) * TH_ + k];
    p = !any;
  } else if (t < 2 * N_TOK) {
    int any = 0;
    for (int k = 0; k < TF_; ++k) any |= fm[((long)b * N_TOK + t - N_TOK) * TF_ + k];
    p = !any;
  } else {
    p = lp[b * M_TOK + t - 2 * N_TOK] != 0;
  }
  pad[i] = p;
}

// pos-MLP layer 1: K=4 -> VALU; writes f16 (feeds a WMMA GEMM)
__global__ void pos_mlp1_kernel(const float* __restrict__ pos, const float* __restrict__ W1,
                                const float* __restrict__ b1, _Float16* __restrict__ h)
{
  long i = (long)blockIdx.x * blockDim.x + threadIdx.x;
  if (i >= (long)B_ * T_ * C_) return;
  int c = (int)(i % C_); long bt = i / C_;
  const float* p = pos + bt * 4;
  float v = b1[c];
#pragma unroll
  for (int k = 0; k < 4; ++k) v += p[k] * W1[c * 4 + k];
  h[i] = (_Float16)(0.5f * v * (1.0f + erff(v * 0.70710678118f)));
}

__global__ void assemble_dec_kernel(const float* __restrict__ tok,
    const int* __restrict__ hk, const int* __restrict__ fk, const int* __restrict__ lk,
    const float* __restrict__ mh, const float* __restrict__ mf, const float* __restrict__ ml,
    const float* __restrict__ posout, const float* __restrict__ types, float* __restrict__ xd)
{
  long i = (long)blockIdx.x * blockDim.x + threadIdx.x;
  if (i >= (long)B_ * T_ * C_) return;
  int c = (int)(i % C_); long bt = i / C_;
  int t = (int)(bt % T_); int b = (int)(bt / T_);
  int keep; const float* mv;
  if (t < N_TOK)          { keep = hk[b * N_TOK + t];             mv = mh; }
  else if (t < 2 * N_TOK) { keep = fk[b * N_TOK + t - N_TOK];     mv = mf; }
  else                    { keep = lk[b * M_TOK + t - 2 * N_TOK]; mv = ml; }
  float v = keep ? tok[i] : mv[c];
  xd[i] = v + posout[i] + types[i];
}

// ---------------- host orchestration ----------------
extern "C" void kernel_launch(void* const* d_in, const int* in_sizes, int n_in,
                              void* d_out, int out_size, void* d_ws, size_t ws_size,
                              hipStream_t stream)
{
  (void)in_sizes; (void)n_in; (void)out_size; (void)ws_size;
  const float* hist   = (const float*)d_in[0];
  const float* lanef  = (const float*)d_in[1];
  const float* fut    = (const float*)d_in[2];
  const int*   hmask  = (const int*)d_in[3];
  const int*   fmask  = (const int*)d_in[4];
  const int*   lpad   = (const int*)d_in[5];
  const float* pos    = (const float*)d_in[6];
  const float* types  = (const float*)d_in[7];
  const int*   hk     = (const int*)d_in[8];
  const int*   fk     = (const int*)d_in[9];
  const int*   lk     = (const int*)d_in[10];
  const int PB = 11;                       // params follow, 11 tensors per block
  auto P = [&](int i) { return (const float*)d_in[i]; };
  const int PNORM = PB + 88;               // norm.g,b dnorm.g,b
  const int PDE   = PNORM + 4;             // dec_embed_W, dec_embed_b
  const int PPOS  = PDE + 2;               // dec_pos W1,b1,W2,b2
  const int PMASK = PPOS + 4;              // mask_hist, mask_fut, mask_lane
  const int PHEAD = PMASK + 3;             // lane_pred_W,b hist_loc_W,b fut_loc_W,b

  const size_t SZ = (size_t)B_ * T_ * C_;  // 6,291,456
  float*     X    = (float*)d_ws;          // residual stream  (f32)
  float*     TOK  = X + SZ;                // decoder-embed out (f32)
  float*     POS2 = TOK + SZ;              // pos-MLP out       (f32)
  _Float16*  LNh  = (_Float16*)(POS2 + SZ);
  _Float16*  QKVh = LNh + SZ;              // 3*SZ
  _Float16*  ATTh = QKVh + 3 * SZ;
  _Float16*  H1h  = ATTh + SZ;             // 4*SZ (also pos-MLP hidden)
  _Float16*  WA   = H1h + 4 * SZ;          // f16 weight arena (~6.8M halves)
  int* encpad = (int*)(WA + 7011328);
  int* decpad = encpad + B_ * T_;

  const long totBTC = (long)B_ * T_ * C_;
  const int EW  = (int)((totBTC + 255) / 256);
  const int PW  = (B_ * T_ + 255) / 256;
  const int LNG = (B_ * T_ * 32 + 255) / 256;
  const int ATG = B_ * H_;                 // one workgroup per (b, h)

  // convert weights to f16 once per launch
  long woff = 0;
  auto cvt = [&](int pidx, long n) {
    _Float16* dst = WA + woff; woff += n;
    cvt_f16_kernel<<<(int)((n + 255) / 256), 256, 0, stream>>>(P(pidx), dst, n);
    return dst;
  };
  _Float16 *Wqkv_h[8], *Wo_h[8], *W1_h[8], *W2_h[8];
  for (int i = 0; i < 8; ++i) {
    int pb = PB + i * 11;
    Wqkv_h[i] = cvt(pb + 2, (long)C3 * C_);
    Wo_h[i]   = cvt(pb + 3, (long)C_ * C_);
    W1_h[i]   = cvt(pb + 7, (long)CM * C_);
    W2_h[i]   = cvt(pb + 9, (long)C_ * CM);
  }
  _Float16* dece_h = cvt(PDE,       (long)C_ * C_);
  _Float16* posw_h = cvt(PPOS + 2,  (long)C_ * C_);
  _Float16* lph_h  = cvt(PHEAD + 0, (long)40 * C_);
  _Float16* xlh_h  = cvt(PHEAD + 2, (long)600 * C_);
  _Float16* ylh_h  = cvt(PHEAD + 4, (long)720 * C_);

  auto gemm_f32 = [&](const _Float16* A, const _Float16* W, const float* bias,
                      const float* resid, float* Out, int N, int K,
                      int a_tok0, int tokcount, long obs, int ooff, int oldo, int act) {
    int ntn = (N + 31) / 32, nblkn = (ntn + 3) / 4;
    int blocks = (B_ * tokcount / 32) * nblkn;
    gemm_wmma_kernel<float><<<blocks, 128, 32 * K * sizeof(_Float16), stream>>>(
        A, W, bias, resid, Out, N, K, T_, a_tok0, tokcount, obs, ooff, oldo, act);
  };
  auto gemm_f16 = [&](const _Float16* A, const _Float16* W, const float* bias,
                      _Float16* Out, int N, int K, int tokcount,
                      long obs, int oldo, int act) {
    int ntn = (N + 31) / 32, nblkn = (ntn + 3) / 4;
    int blocks = (B_ * tokcount / 32) * nblkn;
    gemm_wmma_kernel<_Float16><<<blocks, 128, 32 * K * sizeof(_Float16), stream>>>(
        A, W, bias, nullptr, Out, N, K, T_, 0, tokcount, obs, 0, oldo, act);
  };

  auto block = [&](int pbase, int bi, const int* pad) {
    const float *g1 = P(pbase+0), *b1 = P(pbase+1), *bo = P(pbase+4),
                *g2 = P(pbase+5), *b2 = P(pbase+6), *b1m = P(pbase+8),
                *b2m = P(pbase+10);
    layernorm_kernel<<<LNG, 256, 0, stream>>>(X, g1, b1, LNh, B_ * T_);
    gemm_f16(LNh, Wqkv_h[bi], nullptr, QKVh, C3, C_, T_, (long)T_ * C3, C3, 0);
    attn_kernel<<<ATG, 256, 0, stream>>>(QKVh, pad, ATTh);
    gemm_f32(ATTh, Wo_h[bi], bo, X, X, C_, C_, 0, T_, (long)T_ * C_, 0, C_, 0);
    layernorm_kernel<<<LNG, 256, 0, stream>>>(X, g2, b2, LNh, B_ * T_);
    gemm_f16(LNh, W1_h[bi], b1m, H1h, CM, C_, T_, (long)T_ * CM, CM, 1);   // GELU
    gemm_f32(H1h, W2_h[bi], b2m, X, X, C_, CM, 0, T_, (long)T_ * C_, 0, C_, 0);
  };

  // ---- encoder ----
  build_x_kernel<<<EW, 256, 0, stream>>>(hist, fut, lanef, X);
  enc_pad_kernel<<<PW, 256, 0, stream>>>(hk, fk, lk, encpad);
  for (int i = 0; i < 4; ++i) block(PB + i * 11, i, encpad);
  layernorm_kernel<<<LNG, 256, 0, stream>>>(X, P(PNORM), P(PNORM+1), LNh, B_ * T_);
  gemm_f32(LNh, dece_h, P(PDE+1), nullptr, TOK, C_, C_, 0, T_, (long)T_ * C_, 0, C_, 0);

  // ---- decoder input assembly ----
  pos_mlp1_kernel<<<EW, 256, 0, stream>>>(pos, P(PPOS), P(PPOS+1), H1h);
  gemm_f32(H1h, posw_h, P(PPOS+3), nullptr, POS2, C_, C_, 0, T_, (long)T_ * C_, 0, C_, 0);
  assemble_dec_kernel<<<EW, 256, 0, stream>>>(TOK, hk, fk, lk,
      P(PMASK), P(PMASK+1), P(PMASK+2), POS2, types, X);
  dec_pad_kernel<<<PW, 256, 0, stream>>>(hmask, fmask, lpad, decpad);

  // ---- decoder ----
  for (int i = 0; i < 4; ++i) block(PB + 44 + i * 11, 4 + i, decpad);
  layernorm_kernel<<<LNG, 256, 0, stream>>>(X, P(PNORM+2), P(PNORM+3), LNh, B_ * T_);

  // ---- heads straight into d_out at concat offsets ----
  float* out = (float*)d_out;
  gemm_f32(LNh, lph_h, P(PHEAD+1), nullptr, out, 40,  C_, 2 * N_TOK, M_TOK,
           (long)OUT_PER_B, 0, 40, 0);
  gemm_f32(LNh, xlh_h, P(PHEAD+3), nullptr, out, 600, C_, 0, N_TOK,
           (long)OUT_PER_B, M_TOK * 40, 600, 0);
  gemm_f32(LNh, ylh_h, P(PHEAD+5), nullptr, out, 720, C_, N_TOK, N_TOK,
           (long)OUT_PER_B, M_TOK * 40 + N_TOK * MODES_ * TH_ * 2, 720, 0);
}
